// NonLocal_1614907703717
// MI455X (gfx1250) — matrix operand order repaired and checked by reading, not compiled, not run on hardware
//
#include <hip/hip_runtime.h>
#include <hip/hip_bf16.h>
#include <cstddef>

#define CB 16
#define CC 256
#define CT 64
#define CV 25
#define CN 1600   // CT*CV

typedef __attribute__((ext_vector_type(16))) _Float16 v16h;
typedef __attribute__((ext_vector_type(8)))  _Float16 v8h;
typedef __attribute__((ext_vector_type(8)))  float    v8f;

union V16U { v16h v; v8h h[2]; };

__device__ __forceinline__ v8f wmma32(v16h a, v16h b, v8f c) {
  // D = A(16x32 f16) * B(32x16 f16) + C(16x16 f32)
  return __builtin_amdgcn_wmma_f32_16x16x32_f16(false, a, false, b, (short)0, c, false, false);
}

// -------- Tensor Data Mover support (guarded; falls back to vector copies) --
#if defined(__has_builtin)
#if __has_builtin(__builtin_amdgcn_tensor_load_to_lds) && \
    __has_builtin(__builtin_amdgcn_s_wait_tensorcnt)
#define HAVE_TDM 1
#endif
#endif
#ifndef HAVE_TDM
#define HAVE_TDM 0
#endif

#if HAVE_TDM
typedef __attribute__((ext_vector_type(4))) unsigned int u32x4;
typedef __attribute__((ext_vector_type(8))) int i32x8;
typedef __attribute__((ext_vector_type(4))) int i32x4;

// Issue one TDM tile load: data_size=8B elements, 2D tile
//   tile_d0 units per row, tile_d1 rows (0 => 1-D), row stride stride0 units.
__device__ __forceinline__ void tdm_load_tile(unsigned lds_addr, const void* gptr,
                                              unsigned tile_d0, unsigned tile_d1,
                                              unsigned stride0) {
  const unsigned long long ga = (unsigned long long)(size_t)gptr;
  // group0: count=1 | lds_addr | global_addr[56:0] | type=2
  u32x4 g0 = { 1u, lds_addr, (unsigned)ga,
               (unsigned)(((ga >> 32) & 0x1FFFFFFull) | (2u << 30)) };
  // group1: data_size=3 (8B); tensor_dim0 = tensor_dim1 = 1<<20 (avoid OOB
  // clipping: global_addr is tile start); tile dims + dim0 stride.
  i32x8 g1 = { (int)(3u << 16),
               0,                               // tensor_dim0[15:0] << 16 (=0)
               (int)0x10,                       // td0 hi16 | td1 lo16<<16
               (int)(0x10u | (tile_d0 << 16)),  // td1 hi16 | tile_dim0
               (int)(tile_d1 & 0xFFFFu),        // tile_dim1 | tile_dim2=0
               (int)stride0,                    // tensor_dim0_stride[31:0]
               0, 0 };
  i32x4 gz = { 0, 0, 0, 0 };
#if __clang_major__ >= 23
  i32x8 gz8 = { 0, 0, 0, 0, 0, 0, 0, 0 };
  __builtin_amdgcn_tensor_load_to_lds(g0, g1, gz, gz, gz8, 0);
#else
  __builtin_amdgcn_tensor_load_to_lds(g0, g1, gz, gz, 0);
#endif
}
#endif

// ---- pack x: (B,C,N) f32 -> (B,N,C) f16, LDS-tiled transpose (coalesced) ----
__global__ __launch_bounds__(256) void pack_x_kernel(const float* __restrict__ x,
                                                     _Float16* __restrict__ X16) {
  __shared__ float t[32][33];
  const int bz = blockIdx.x;             // b*400 + nt*8 + ct
  const int b  = bz / 400;
  const int rem = bz % 400;
  const int n0 = (rem / 8) * 32;
  const int c0 = (rem % 8) * 32;
  const int tx = threadIdx.x & 31;
  const int ty = threadIdx.x >> 5;       // 0..7
#pragma unroll
  for (int i = 0; i < 4; ++i) {
    const int cl = ty + i * 8;
    t[cl][tx] = x[(size_t)b * CC * CN + (size_t)(c0 + cl) * CN + n0 + tx];
  }
  __syncthreads();
#pragma unroll
  for (int i = 0; i < 4; ++i) {
    const int nl = ty + i * 8;
    X16[(size_t)b * CN * CC + (size_t)(n0 + nl) * CC + c0 + tx] =
        (_Float16)t[tx][nl];
  }
}

// ---- pack W1..W3 -> f16 (row major; B-operand wants W[col][k]) ----
__global__ __launch_bounds__(256) void pack_w_kernel(const float* __restrict__ W1,
                                                     const float* __restrict__ W2,
                                                     const float* __restrict__ W3,
                                                     _Float16* __restrict__ W16) {
  int idx = blockIdx.x * 256 + threadIdx.x;      // < 3*C*C
  int w = idx / (CC * CC);
  int off = idx - w * CC * CC;
  const float* src = (w == 0) ? W1 : (w == 1) ? W2 : W3;
  W16[idx] = (_Float16)src[off];
}

// ---- QKV projection GEMM: (B*N x C) @ (C x C) + bias --------------------
__global__ __launch_bounds__(256) void qkv_kernel(
    const _Float16* __restrict__ X16, const _Float16* __restrict__ W16,
    const float* __restrict__ b1, const float* __restrict__ b2,
    const float* __restrict__ b3,
    _Float16* __restrict__ Q16, _Float16* __restrict__ K16,
    _Float16* __restrict__ V16) {
  const int lane = threadIdx.x & 31;
  const int wave = threadIdx.x >> 5;
  const int row0 = blockIdx.x * 128 + wave * 16;
  const int wsel = blockIdx.y;
  const _Float16* Wm = W16 + (size_t)wsel * CC * CC;
  const float* bias = (wsel == 0) ? b1 : (wsel == 1) ? b2 : b3;
  _Float16* out = (wsel == 0) ? Q16 : (wsel == 1) ? K16 : V16;

  const int mrow = row0 + (lane & 15);
  const int hs = (lane & 16) ? 16 : 0;
  V16U a[8];
#pragma unroll
  for (int cc = 0; cc < 8; ++cc) {
    const v8h* p = (const v8h*)(X16 + (size_t)mrow * CC + cc * 32 + hs);
    a[cc].h[0] = p[0];    // k = hs..hs+7
    a[cc].h[1] = p[1];    // k = hs+8..hs+15
  }
  const int kb = (lane & 16) ? 16 : 0;
  const int rbase = row0 + ((lane & 16) ? 8 : 0);
#pragma unroll
  for (int ct = 0; ct < 16; ++ct) {
    const int col = ct * 16 + (lane & 15);
    v8f acc = {};
#pragma unroll
    for (int cc = 0; cc < 8; ++cc) {
      V16U bf;
      const v8h* p = (const v8h*)(Wm + (size_t)col * CC + cc * 32 + kb);
      bf.h[0] = p[0];
      bf.h[1] = p[1];
      acc = wmma32(a[cc].v, bf.v, acc);
    }
    const float bb = bias[col];
#pragma unroll
    for (int r = 0; r < 8; ++r)
      out[(size_t)(rbase + r) * CC + col] = (_Float16)(acc[r] + bb);
  }
}

// ---- V (B,N,C) -> VT (B,C,N), f16 LDS-tiled transpose (coalesced) --------
__global__ __launch_bounds__(256) void vt_kernel(const _Float16* __restrict__ V16,
                                                 _Float16* __restrict__ VT16) {
  __shared__ _Float16 t[32][33];
  const int bz = blockIdx.x;
  const int b  = bz / 400;
  const int rem = bz % 400;
  const int n0 = (rem / 8) * 32;
  const int c0 = (rem % 8) * 32;
  const int tx = threadIdx.x & 31;
  const int ty = threadIdx.x >> 5;
#pragma unroll
  for (int i = 0; i < 4; ++i) {
    const int nl = ty + i * 8;
    t[nl][tx] = V16[(size_t)b * CN * CC + (size_t)(n0 + nl) * CC + c0 + tx];
  }
  __syncthreads();
#pragma unroll
  for (int i = 0; i < 4; ++i) {
    const int cl = ty + i * 8;
    VT16[(size_t)b * CC * CN + (size_t)(c0 + cl) * CN + n0 + tx] = t[tx][cl];
  }
}

// ---- residual path: R[b,n] = sum_c x[b,c,n] * W4[C+c]  (coalesced in n) ---
__global__ __launch_bounds__(256) void xw4_kernel(const float* __restrict__ x,
                                                  const float* __restrict__ W4,
                                                  float* __restrict__ R) {
  const int idx = blockIdx.x * 256 + threadIdx.x;   // < B*N
  const int b = idx / CN;
  const int n = idx - b * CN;
  const float* xb = x + (size_t)b * CC * CN + n;
  float s = 0.0f;
#pragma unroll 8
  for (int c = 0; c < CC; ++c) s += xb[(size_t)c * CN] * W4[CC + c];
  R[idx] = s;
}

// ---- flash attention: per wave 16 query rows, 32-key tiles, TDM staging ---
__global__ __launch_bounds__(128) void attn_kernel(
    const _Float16* __restrict__ Q16, const _Float16* __restrict__ K16,
    const _Float16* __restrict__ VT16, float* __restrict__ CTX) {
  __shared__ _Float16 Kl[2][32 * CC];     // 2 x 16 KB: K tile [key][c]
  __shared__ _Float16 Vl[2][CC * 32];     // 2 x 16 KB: VT tile [c][key]
  __shared__ _Float16 Pl[4 * 16 * 32];    //     4 KB: per-wave P buffer

  const int tid  = threadIdx.x;
  const int lane = tid & 31;
  const int wave = tid >> 5;
  const int b    = blockIdx.x / 25;
  const int tile = blockIdx.x % 25;
  const int m0   = tile * 64 + wave * 16;
  const size_t rowbase = (size_t)b * CN + m0;
  const size_t vtbase  = (size_t)b * CC * CN;

  // Q fragments for the whole row tile (16x256 f16)
  const int hs = (lane & 16) ? 16 : 0;
  const int mrow = (int)rowbase + (lane & 15);
  V16U qa[8];
#pragma unroll
  for (int cc = 0; cc < 8; ++cc) {
    const v8h* p = (const v8h*)(Q16 + (size_t)mrow * CC + cc * 32 + hs);
    qa[cc].h[0] = p[0];
    qa[cc].h[1] = p[1];
  }

  v8f zero = {};
  v8f o[16];
#pragma unroll
  for (int ct = 0; ct < 16; ++ct) o[ct] = zero;
  float mr_[8], lr_[8];
#pragma unroll
  for (int r = 0; r < 8; ++r) { mr_[r] = -1e30f; lr_[r] = 0.0f; }

  const int kb = (lane & 16) ? 16 : 0;
  const int fr = tid >> 2;            // 0..31  K staging row (fallback)
  const int fc = (tid & 3) * 64;      // K staging col base  (fallback)

#if HAVE_TDM
  // prologue: kick off tile 0 into buffer 0 (one wave issues; TENSORcnt)
  if (wave == 0) {
    tdm_load_tile((unsigned)(size_t)&Kl[0][0],
                  K16 + ((size_t)b * CN + 0) * CC, 2048u, 0u, 2048u);
    tdm_load_tile((unsigned)(size_t)&Vl[0][0],
                  VT16 + vtbase + 0, 8u, 256u, 400u);
  }
#endif

  for (int jj = 0, it = 0; jj < CN; jj += 32, ++it) {
    const int buf = it & 1;
    __syncthreads();                  // everyone done reading buf (2 iters ago)
#if HAVE_TDM
    if (wave == 0) {
      if (jj + 32 < CN) {
        // overlap: DMA the next key tile into the other buffer
        tdm_load_tile((unsigned)(size_t)&Kl[buf ^ 1][0],
                      K16 + ((size_t)b * CN + jj + 32) * CC, 2048u, 0u, 2048u);
        tdm_load_tile((unsigned)(size_t)&Vl[buf ^ 1][0],
                      VT16 + vtbase + jj + 32, 8u, 256u, 400u);
        __builtin_amdgcn_s_wait_tensorcnt(2);   // current buf's 2 loads done
      } else {
        __builtin_amdgcn_s_wait_tensorcnt(0);
      }
    }
#else
    {
      // stage K tile (32 keys x 256 ch), all-vector copies
      const size_t krow = (size_t)b * CN + jj + fr;
      const v8h* gk = (const v8h*)(K16 + krow * CC + fc);
#pragma unroll
      for (int i = 0; i < 8; ++i) *(v8h*)&Kl[buf][fr * CC + fc + i * 8] = gk[i];
      // stage VT tile (256 ch x 32 keys), all-vector copies
#pragma unroll
      for (int i = 0; i < 2; ++i) {
        const int c = tid * 2 + i;
        const v8h* gv = (const v8h*)(VT16 + vtbase + (size_t)c * CN + jj);
        v8h* dst = (v8h*)&Vl[buf][c * 32];
        dst[0] = gv[0]; dst[1] = gv[1]; dst[2] = gv[2]; dst[3] = gv[3];
      }
      if (jj + 32 < CN) {
        __builtin_prefetch(K16 + (krow + 32) * CC + fc, 0, 1);
        __builtin_prefetch(VT16 + vtbase + (size_t)(tid * 2) * CN + jj + 32, 0, 1);
      }
    }
#endif
    __syncthreads();

    // S = Q * K^T : 16x32 scores as two 16x16 f32 fragments
    v8f S0 = zero, S1 = zero;
#pragma unroll
    for (int cc = 0; cc < 8; ++cc) {
      V16U bf0, bf1;
      const v8h* p0 = (const v8h*)&Kl[buf][(lane & 15) * CC + cc * 32 + kb];
      bf0.h[0] = p0[0]; bf0.h[1] = p0[1];
      const v8h* p1 = (const v8h*)&Kl[buf][(16 + (lane & 15)) * CC + cc * 32 + kb];
      bf1.h[0] = p1[0]; bf1.h[1] = p1[1];
      S0 = wmma32(qa[cc].v, bf0.v, S0);
      S1 = wmma32(qa[cc].v, bf1.v, S1);
    }

    // online softmax; a row's 16 cols live across 16 lanes of a half-wave
    const int prow = (lane & 16) ? 8 : 0;
#pragma unroll
    for (int r = 0; r < 8; ++r) {
      float cur = fmaxf(S0[r], S1[r]);
#pragma unroll
      for (int off = 1; off < 16; off <<= 1)
        cur = fmaxf(cur, __shfl_xor(cur, off, 16));
      float mn = fmaxf(mr_[r], cur);
      float sc = __expf(mr_[r] - mn);
      float p0 = __expf(S0[r] - mn);
      float p1 = __expf(S1[r] - mn);
      float ps = p0 + p1;
#pragma unroll
      for (int off = 1; off < 16; off <<= 1) ps += __shfl_xor(ps, off, 16);
      lr_[r] = lr_[r] * sc + ps;
      mr_[r] = mn;
      Pl[wave * 512 + (r + prow) * 32 + (lane & 15)]      = (_Float16)p0;
      Pl[wave * 512 + (r + prow) * 32 + 16 + (lane & 15)] = (_Float16)p1;
#pragma unroll
      for (int ct = 0; ct < 16; ++ct) o[ct][r] *= sc;
    }

    // reload P as A-fragment (C-layout -> A-layout via per-wave LDS buffer)
    V16U pa;
    const v8h* pp = (const v8h*)&Pl[wave * 512 + (lane & 15) * 32 + hs];
    pa.h[0] = pp[0];
  pa.h[1] = pp[1];

    // O += P * V ; VT B-fragments are contiguous 16-byte LDS loads
#pragma unroll
    for (int ct = 0; ct < 16; ++ct) {
      V16U bf;
      const v8h* p = (const v8h*)&Vl[buf][(ct * 16 + (lane & 15)) * 32 + kb];
      bf.h[0] = p[0]; bf.h[1] = p[1];
      o[ct] = wmma32(pa.v, bf.v, o[ct]);
    }
  }

  // normalize and write ctx (f32)
  float invl[8];
#pragma unroll
  for (int r = 0; r < 8; ++r) invl[r] = 1.0f / lr_[r];
  const size_t rb2 = rowbase + ((lane & 16) ? 8 : 0);
#pragma unroll
  for (int ct = 0; ct < 16; ++ct)
#pragma unroll
    for (int r = 0; r < 8; ++r)
      CTX[(rb2 + r) * CC + ct * 16 + (lane & 15)] = o[ct][r] * invl[r];
}

// ---- head: out = ctx.w4a + R + b4 (one wave per output, coalesced) -------
__global__ __launch_bounds__(256) void head_kernel(
    const float* __restrict__ CTX, const float* __restrict__ R,
    const float* __restrict__ W4, const float* __restrict__ b4,
    float* __restrict__ out) {
  const int lane = threadIdx.x & 31;
  const int wave = threadIdx.x >> 5;
  const int idx = blockIdx.x * 8 + wave;   // < B*N
  float s = 0.0f;
#pragma unroll
  for (int i = 0; i < 8; ++i) {
    const int c = lane + 32 * i;
    s += CTX[(size_t)idx * CC + c] * W4[c];
  }
#pragma unroll
  for (int off = 16; off >= 1; off >>= 1) s += __shfl_xor(s, off, 32);
  if (lane == 0) out[idx] = s + R[idx] + b4[0];
}

extern "C" void kernel_launch(void* const* d_in, const int* in_sizes, int n_in,
                              void* d_out, int out_size, void* d_ws, size_t ws_size,
                              hipStream_t stream) {
  const float* x  = (const float*)d_in[0];
  const float* W1 = (const float*)d_in[1];
  const float* b1 = (const float*)d_in[2];
  const float* W2 = (const float*)d_in[3];
  const float* b2 = (const float*)d_in[4];
  const float* W3 = (const float*)d_in[5];
  const float* b3 = (const float*)d_in[6];
  const float* W4 = (const float*)d_in[7];
  const float* b4 = (const float*)d_in[8];
  float* out = (float*)d_out;
  (void)in_sizes; (void)n_in; (void)out_size; (void)ws_size;

  char* ws = (char*)d_ws;
  const size_t xhalf = (size_t)CB * CN * CC * sizeof(_Float16); // 13,107,200 B
  size_t off = 0;
  _Float16* X16  = (_Float16*)(ws + off); off += xhalf;
  _Float16* W16  = (_Float16*)(ws + off); off += (size_t)3 * CC * CC * sizeof(_Float16);
  _Float16* Q16  = (_Float16*)(ws + off); off += xhalf;
  _Float16* K16  = (_Float16*)(ws + off); off += xhalf;
  _Float16* V16  = (_Float16*)(ws + off); off += xhalf;
  _Float16* VT16 = (_Float16*)(ws + off); off += xhalf;
  float*    Rres = (float*)(ws + off);    off += (size_t)CB * CN * sizeof(float);
  float*    CTX  = (float*)(ws + off);    off += (size_t)CB * CN * CC * sizeof(float);

  pack_x_kernel<<<CB * 400, 256, 0, stream>>>(x, X16);
  pack_w_kernel<<<(3 * CC * CC) / 256, 256, 0, stream>>>(W1, W2, W3, W16);
  xw4_kernel<<<(CB * CN) / 256, 256, 0, stream>>>(x, W4, Rres);
  qkv_kernel<<<dim3((CB * CN) / 128, 3), 256, 0, stream>>>(
      X16, W16, b1, b2, b3, Q16, K16, V16);
  vt_kernel<<<CB * 400, 256, 0, stream>>>(V16, VT16);
  attn_kernel<<<CB * (CN / 64), 128, 0, stream>>>(Q16, K16, VT16, CTX);
  head_kernel<<<(CB * CN) / 8, 256, 0, stream>>>(CTX, Rres, W4, b4, out);
}
